// MInferModel_12275016532524
// MI455X (gfx1250) — compile-verified, matrix operand order checked by reading, not run on hardware
//
#include <hip/hip_runtime.h>

#define S_LEN   4096
#define DHEAD   128
#define NHEAD   8
#define LASTQ   64
#define V_TOPK_N 1024
#define S_TOPK_N 2048
#define FORCE_V_N 30
#define FORCE_S_N 100
#define SCALE 0.08838834764831845f   // 1/sqrt(128)

typedef __bf16 bf16;
typedef __attribute__((ext_vector_type(16))) __bf16 v16bf;
typedef __attribute__((ext_vector_type(8)))  float  v8f;
typedef __attribute__((ext_vector_type(4)))  float  f32x4;
typedef __attribute__((ext_vector_type(4)))  __bf16 bf16x4;

// K-index striping for 16-bit A/B fragments of v_wmma_*_16x16x32_bf16:
// VGPR v holds K pairs (ISA 7.12.2); lanes 16-31 offset by +8 in each group.
__device__ __forceinline__ int kmap(int e, int hi) {
  int t = e >> 1;
  int k = (t < 4) ? (2 * t) : (16 + 2 * (t - 4));
  return k + (e & 1) + (hi ? 8 : 0);
}

__device__ __forceinline__ v8f wmma_bf16(v16bf a, v16bf b, v8f c) {
  return __builtin_amdgcn_wmma_f32_16x16x32_bf16(false, a, false, b, (short)0, c,
                                                 false, false);
}

__device__ __forceinline__ v16bf load_afrag_global(const float* rowPtr, int kbase, int hi) {
  v16bf a;
#pragma unroll
  for (int e = 0; e < 16; ++e) a[e] = (bf16)rowPtr[kbase + kmap(e, hi)];
  return a;
}

__device__ __forceinline__ v16bf load_bfrag_k(const bf16* ktile, int keyLocal,
                                              int kbase, int hi) {
  v16bf b;
#pragma unroll
  for (int e = 0; e < 16; ++e)
    b[e] = ktile[keyLocal * DHEAD + kbase + kmap(e, hi)];
  return b;
}

__device__ __forceinline__ v16bf load_bfrag_v(const bf16* vtile, int dcol,
                                              int kbase, int hi) {
  v16bf b;
#pragma unroll
  for (int e = 0; e < 16; ++e)
    b[e] = vtile[(kbase + kmap(e, hi)) * DHEAD + dcol];
  return b;
}

__device__ __forceinline__ v16bf load_afrag_p(const bf16* ptile, int row,
                                              int kbase, int hi) {
  v16bf a;
#pragma unroll
  for (int e = 0; e < 16; ++e)
    a[e] = ptile[row * 64 + kbase + kmap(e, hi)];
  return a;
}

// ---- 16-lane-half xor exchange: VALU permlane16 (no LDS pipe, no dscnt)
__device__ __forceinline__ float lanexor16(float v, int w) {
#if __has_builtin(__builtin_amdgcn_permlane16)
  int iv = __float_as_int(v);
  int r;
  switch (w) {
    case 0:  r = __builtin_amdgcn_permlane16(iv, iv, 0x67452301u, 0xEFCDAB89u, false, false); break;
    case 1:  r = __builtin_amdgcn_permlane16(iv, iv, 0x54761032u, 0xDCFE98BAu, false, false); break;
    case 2:  r = __builtin_amdgcn_permlane16(iv, iv, 0x32107654u, 0xBA98FEDCu, false, false); break;
    default: r = __builtin_amdgcn_permlane16(iv, iv, 0xFEDCBA98u, 0x76543210u, false, false); break;
  }
  return __int_as_float(r);
#else
  return __shfl_xor(v, 1 << w, 32);
#endif
}

__device__ __forceinline__ void halfmax8(float v[8]) {
#pragma unroll
  for (int w = 0; w < 4; ++w) {
    float t[8];
#pragma unroll
    for (int e = 0; e < 8; ++e) t[e] = lanexor16(v[e], w);
#pragma unroll
    for (int e = 0; e < 8; ++e) v[e] = fmaxf(v[e], t[e]);
  }
}
__device__ __forceinline__ void halfsum8(float v[8]) {
#pragma unroll
  for (int w = 0; w < 4; ++w) {
    float t[8];
#pragma unroll
    for (int e = 0; e < 8; ++e) t[e] = lanexor16(v[e], w);
#pragma unroll
    for (int e = 0; e < 8; ++e) v[e] += t[e];
  }
}

// largest threshold t (uint bits of a non-negative float) with count(>=t) >= K
__device__ unsigned select_threshold(const float* vals, int K, int tid, int* cnt) {
  unsigned t = 0;
  for (int bit = 31; bit >= 0; --bit) {
    unsigned cand = t | (1u << bit);
    if (tid == 0) *cnt = 0;
    __syncthreads();
    int c = 0;
    for (int i = tid; i < S_LEN; i += 128)
      if (__float_as_uint(vals[i]) >= cand) ++c;
    atomicAdd(cnt, c);
    __syncthreads();
    if (*cnt >= K) t = cand;
    __syncthreads();
  }
  return t;
}

// f32x4 -> packed bf16x4 (v_cvt_pk_bf16_f32 x2)
__device__ __forceinline__ bf16x4 cvt4(f32x4 a) {
  bf16x4 c;
  c.x = (bf16)a.x; c.y = (bf16)a.y; c.z = (bf16)a.z; c.w = (bf16)a.w;
  return c;
}

// mask + branchless online-softmax update for one 16-row tile (C layout)
__device__ __forceinline__ void online_update(
    v8f sc[4], float* m, float* l, v8f* o, bf16* pw, int i0, int jbase,
    const unsigned char* vm, const unsigned char* sm, int n, int hi) {
#pragma unroll
  for (int e = 0; e < 8; ++e) {
    int i = i0 + e;
#pragma unroll
    for (int ct = 0; ct < 4; ++ct) {
      int j = jbase + ct * 16 + n;
      float s = -3.0e38f;
      if (j <= i) {
        int d = i - j;
        if (vm[j] | sm[d]) s = sc[ct][e] * SCALE;
      }
      sc[ct][e] = s;
    }
  }
  float rmax[8];
#pragma unroll
  for (int e = 0; e < 8; ++e)
    rmax[e] = fmaxf(fmaxf(sc[0][e], sc[1][e]), fmaxf(sc[2][e], sc[3][e]));
  halfmax8(rmax);

  float mn[8], rsum[8];
#pragma unroll
  for (int e = 0; e < 8; ++e) {
    mn[e] = fmaxf(m[e], rmax[e]);
    float rs = 0.f;
#pragma unroll
    for (int ct = 0; ct < 4; ++ct) {
      float p = (sc[ct][e] > -1.0e37f) ? __expf(sc[ct][e] - mn[e]) : 0.f;
      sc[ct][e] = p;
      rs += p;
    }
    rsum[e] = rs;
  }
  halfsum8(rsum);

#pragma unroll
  for (int e = 0; e < 8; ++e) {
    float corr = __expf(m[e] - mn[e]);   // ==1 when row untouched so far
    l[e] = l[e] * corr + rsum[e];
    m[e] = mn[e];
#pragma unroll
    for (int t = 0; t < 8; ++t) o[t][e] *= corr;
    int r = e + 8 * hi;
#pragma unroll
    for (int ct = 0; ct < 4; ++ct)
      pw[r * 64 + ct * 16 + n] = (bf16)sc[ct][e];
  }
}

// ---------------------------------------------------------------------------
// Kernel A: pattern estimation for one head per block (128 threads, 4 waves)
// ---------------------------------------------------------------------------
__global__ __launch_bounds__(128) void pattern_kernel(
    const float* __restrict__ q, const float* __restrict__ k,
    unsigned char* __restrict__ vmask, unsigned char* __restrict__ smask) {
  const int h = blockIdx.x;
  const int tid = threadIdx.x;
  const int wave = tid >> 5, lane = tid & 31;
  const int hi = lane >> 4, n = lane & 15;

  __shared__ __align__(16) bf16 ktile[16 * DHEAD];
  __shared__ float vert[S_LEN];
  __shared__ float slsh[S_LEN];
  __shared__ int   s_cnt;

  for (int i = tid; i < S_LEN; i += 128) { vert[i] = 0.f; slsh[i] = 0.f; }

  const float* qh = q + (size_t)h * S_LEN * DHEAD;
  const float* kh = k + (size_t)h * S_LEN * DHEAD;

  const int qrow_lane = S_LEN - LASTQ + wave * 16 + n;
  v16bf aq[4];
#pragma unroll
  for (int ks = 0; ks < 4; ++ks)
    aq[ks] = load_afrag_global(qh + (size_t)qrow_lane * DHEAD, ks * 32, hi);

  float m[8], l[8];
#pragma unroll
  for (int e = 0; e < 8; ++e) { m[e] = -3.0e38f; l[e] = 0.f; }

  const v8f zero8 = {};
  __syncthreads();

  // ---- pass 1: online row max / sumexp over all 256 key tiles ----
  for (int kt = 0; kt < S_LEN / 16; ++kt) {
    const f32x4* kg = (const f32x4*)(kh + (size_t)(kt * 16) * DHEAD);
    for (int idx = tid; idx < 16 * DHEAD / 4; idx += 128)
      ((bf16x4*)ktile)[idx] = cvt4(kg[idx]);
    __syncthreads();

    v16bf bk[4];
#pragma unroll
    for (int ks = 0; ks < 4; ++ks) bk[ks] = load_bfrag_k(ktile, n, ks * 32, hi);
    v8f acc = zero8;
#pragma unroll
    for (int ks = 0; ks < 4; ++ks) acc = wmma_bf16(aq[ks], bk[ks], acc);

    float s8[8], rmax[8];
#pragma unroll
    for (int e = 0; e < 8; ++e) {
      int qp = S_LEN - LASTQ + wave * 16 + e + 8 * hi;
      int j = kt * 16 + n;
      s8[e] = (j <= qp) ? acc[e] * SCALE : -3.0e38f;
      rmax[e] = s8[e];
    }
    halfmax8(rmax);
    float mn[8], rsum[8];
#pragma unroll
    for (int e = 0; e < 8; ++e) {
      mn[e] = fmaxf(m[e], rmax[e]);
      rsum[e] = (s8[e] > -1.0e37f) ? __expf(s8[e] - mn[e]) : 0.f;
    }
    halfsum8(rsum);
#pragma unroll
    for (int e = 0; e < 8; ++e) {
      float corr = __expf(m[e] - mn[e]);
      l[e] = l[e] * corr + rsum[e];
      m[e] = mn[e];
    }
    __syncthreads();
  }

  // ---- pass 2: recompute probs, accumulate vertical / slash scores ----
  for (int kt = 0; kt < S_LEN / 16; ++kt) {
    const f32x4* kg = (const f32x4*)(kh + (size_t)(kt * 16) * DHEAD);
    for (int idx = tid; idx < 16 * DHEAD / 4; idx += 128)
      ((bf16x4*)ktile)[idx] = cvt4(kg[idx]);
    __syncthreads();

    v16bf bk[4];
#pragma unroll
    for (int ks = 0; ks < 4; ++ks) bk[ks] = load_bfrag_k(ktile, n, ks * 32, hi);
    v8f acc = zero8;
#pragma unroll
    for (int ks = 0; ks < 4; ++ks) acc = wmma_bf16(aq[ks], bk[ks], acc);

    float csum = 0.f;
#pragma unroll
    for (int e = 0; e < 8; ++e) {
      int qp = S_LEN - LASTQ + wave * 16 + e + 8 * hi;
      int j = kt * 16 + n;
      if (j <= qp) {
        float p = __expf(acc[e] * SCALE - m[e]) / l[e];
        csum += p;
        atomicAdd(&slsh[qp - j], p);          // ds_add_f32
      }
    }
    atomicAdd(&vert[kt * 16 + n], csum);
    __syncthreads();
  }

  // ---- forced regions + top-k threshold selection + mask emit ----
  for (int i = tid; i < FORCE_V_N; i += 128) vert[i] = 3.0e38f;
  for (int i = tid; i < FORCE_S_N; i += 128) slsh[i] = 3.0e38f;
  __syncthreads();

  unsigned tv = select_threshold(vert, V_TOPK_N, tid, &s_cnt);
  unsigned ts = select_threshold(slsh, S_TOPK_N, tid, &s_cnt);

  for (int i = tid; i < S_LEN; i += 128) {
    vmask[(size_t)h * S_LEN + i] = (__float_as_uint(vert[i]) >= tv) ? 1 : 0;
    smask[(size_t)h * S_LEN + i] = (__float_as_uint(slsh[i]) >= ts) ? 1 : 0;
  }
}

// ---------------------------------------------------------------------------
// Kernel B: masked flash attention. grid = (S/128, H), 128 threads (4 waves),
// each wave owns TWO 16-row query tiles (32 rows) so every K/V B-fragment
// feeds two WMMAs (halved LDS bytes per MAC).
// ---------------------------------------------------------------------------
__global__ __launch_bounds__(128) void vs_attn_kernel(
    const float* __restrict__ q, const float* __restrict__ k,
    const float* __restrict__ v, const unsigned char* __restrict__ vmask,
    const unsigned char* __restrict__ smask, float* __restrict__ out) {
  const int qb = blockIdx.x;   // 0..31
  const int h  = blockIdx.y;   // 0..7
  const int tid = threadIdx.x;
  const int wave = tid >> 5, lane = tid & 31;
  const int hi = lane >> 4, n = lane & 15;
  const int qbase = qb * 128;

  __shared__ __align__(16) bf16 kt_lds[64 * DHEAD];   // 16 KB
  __shared__ __align__(16) bf16 vt_lds[64 * DHEAD];   // 16 KB
  __shared__ __align__(16) bf16 p_lds[4][32 * 64];    // 16 KB (wave-private)
  __shared__ unsigned char vm[S_LEN];                 // 4 KB
  __shared__ unsigned char sm[S_LEN];                 // 4 KB
  __shared__ int blk_flag;

  const float* qh = q + (size_t)h * S_LEN * DHEAD;
  const float* kh = k + (size_t)h * S_LEN * DHEAD;
  const float* vh = v + (size_t)h * S_LEN * DHEAD;

  for (int i = tid; i < S_LEN; i += 128) {
    vm[i] = vmask[(size_t)h * S_LEN + i];
    sm[i] = smask[(size_t)h * S_LEN + i];
  }

  v16bf aq[2][4];
#pragma unroll
  for (int u = 0; u < 2; ++u) {
    int qrow = qbase + wave * 32 + u * 16 + n;
#pragma unroll
    for (int ks = 0; ks < 4; ++ks)
      aq[u][ks] = load_afrag_global(qh + (size_t)qrow * DHEAD, ks * 32, hi);
  }

  const v8f zero8 = {};
  float m[2][8], l[2][8];
  v8f o[2][8];
#pragma unroll
  for (int u = 0; u < 2; ++u) {
#pragma unroll
    for (int e = 0; e < 8; ++e) { m[u][e] = -3.0e38f; l[u][e] = 0.f; }
#pragma unroll
    for (int t = 0; t < 8; ++t) o[u][t] = zero8;
  }

  const int nkb = 2 * qb + 2;
  for (int kb = 0; kb < nkb; ++kb) {
    __syncthreads();                                   // A: close prev iteration
    if (tid == 0) blk_flag = 0;
    __syncthreads();                                   // B
    // ---- key-block skip test: any allowed column or diagonal? ----
    if (tid < 64) {
      if (vm[kb * 64 + tid]) atomicOr(&blk_flag, 1);
    } else {
      int d0 = qbase - kb * 64 - 63;                   // dist range width 191
#pragma unroll
      for (int u = 0; u < 3; ++u) {
        int d = d0 + (tid - 64) * 3 + u;
        if (d >= 0 && d < S_LEN && sm[d]) atomicOr(&blk_flag, 1);
      }
    }
    __syncthreads();                                   // C
    if (!blk_flag) continue;                           // uniform across block

    // ---- stage K/V tiles to LDS as packed bf16 ----
    {
      const f32x4* kg = (const f32x4*)(kh + (size_t)(kb * 64) * DHEAD);
      const f32x4* vg = (const f32x4*)(vh + (size_t)(kb * 64) * DHEAD);
      for (int idx = tid; idx < 64 * DHEAD / 4; idx += 128) {
        ((bf16x4*)kt_lds)[idx] = cvt4(kg[idx]);
        ((bf16x4*)vt_lds)[idx] = cvt4(vg[idx]);
      }
    }
    if (kb + 1 < nkb)
      __builtin_prefetch(kh + (size_t)((kb + 1) * 64) * DHEAD + tid * 8, 0, 3);
    __syncthreads();                                   // D

    // ---- scores: two 16x64 tiles per wave; B-frags reused across u ----
    v8f sc[2][4];
#pragma unroll
    for (int ct = 0; ct < 4; ++ct) {
      v16bf bk[4];
#pragma unroll
      for (int ks = 0; ks < 4; ++ks)
        bk[ks] = load_bfrag_k(kt_lds, ct * 16 + n, ks * 32, hi);
#pragma unroll
      for (int u = 0; u < 2; ++u) {
        v8f a = zero8;
#pragma unroll
        for (int ks = 0; ks < 4; ++ks) a = wmma_bf16(aq[u][ks], bk[ks], a);
        sc[u][ct] = a;
      }
    }

    // ---- mask + online softmax per u-tile ----
    bf16* pw = &p_lds[wave][0];
#pragma unroll
    for (int u = 0; u < 2; ++u)
      online_update(sc[u], m[u], l[u], o[u], pw + u * 16 * 64,
                    qbase + wave * 32 + u * 16 + 8 * hi, kb * 64, vm, sm, n, hi);
    __syncthreads();                                   // E: P visible

    // ---- PV: O += P(32x64) * V(64x128); P A-frags hoisted, V reused ----
    v16bf ap[2][2];
#pragma unroll
    for (int u = 0; u < 2; ++u)
#pragma unroll
      for (int ks2 = 0; ks2 < 2; ++ks2)
        ap[u][ks2] = load_afrag_p(pw + u * 16 * 64, n, ks2 * 32, hi);
#pragma unroll
    for (int t = 0; t < 8; ++t) {
      v16bf bv0 = load_bfrag_v(vt_lds, t * 16 + n, 0, hi);
      v16bf bv1 = load_bfrag_v(vt_lds, t * 16 + n, 32, hi);
#pragma unroll
      for (int u = 0; u < 2; ++u) {
        v8f a = o[u][t];
        a = wmma_bf16(ap[u][0], bv0, a);
        a = wmma_bf16(ap[u][1], bv1, a);
        o[u][t] = a;
      }
    }
  }

  // ---- epilogue: normalize and store ----
#pragma unroll
  for (int u = 0; u < 2; ++u)
#pragma unroll
    for (int e = 0; e < 8; ++e) {
      int i = qbase + wave * 32 + u * 16 + e + 8 * hi;
      float inv = (l[u][e] > 0.f) ? 1.0f / l[u][e] : 0.f;
#pragma unroll
      for (int t = 0; t < 8; ++t)
        out[((size_t)h * S_LEN + i) * DHEAD + t * 16 + n] = o[u][t][e] * inv;
    }
}

extern "C" void kernel_launch(void* const* d_in, const int* in_sizes, int n_in,
                              void* d_out, int out_size, void* d_ws, size_t ws_size,
                              hipStream_t stream) {
  (void)in_sizes; (void)n_in; (void)out_size; (void)ws_size;
  const float* q = (const float*)d_in[0];
  const float* k = (const float*)d_in[1];
  const float* v = (const float*)d_in[2];
  float* out = (float*)d_out;

  unsigned char* vmask = (unsigned char*)d_ws;                 // H*S bytes
  unsigned char* smask = vmask + (size_t)NHEAD * S_LEN;        // H*S bytes

  pattern_kernel<<<dim3(NHEAD), dim3(128), 0, stream>>>(q, k, vmask, smask);
  vs_attn_kernel<<<dim3(S_LEN / 128, NHEAD), dim3(128), 0, stream>>>(
      q, k, v, vmask, smask, out);
}
